// SAGE_37160057045597
// MI455X (gfx1250) — compile-verified, matrix operand order checked by reading, not run on hardware
//
#include <hip/hip_runtime.h>
#include <stdint.h>

// ---------------------------------------------------------------------------
// GraphSAGE 2-layer forward for MI455X (gfx1250, wave32).
//
// Roofline: GEMMs ~5 GFLOP (trivial on WMMA); the limiter is the edge
// scatter (E=640K, d=128). Working set (~65MB) is L2-resident (192MB), so
// gathers + f32 atomics resolve on-die. All f32->f16 conversion lives in
// one-time prep passes (xH, aggH pre-scaled by 1/max(deg,1), h written as
// f16 by layer-1 epilogue), so gathers move 256B/edge and the WMMA loop is
// pure b128 loads + v_wmma_f32_16x16x32_f16.
//
// GEMM structure: B fragments (f16 transposed weights) hoisted into 64
// VGPRs and reused across 5 M-tiles; per tile, ALL 16 A-side b128 loads are
// issued into distinct registers before the 8 chained WMMAs, letting the
// compiler clause the loads and use partial s_wait_loadcnt to overlap
// fetch latency with WMMA issue.
// ---------------------------------------------------------------------------

typedef __attribute__((ext_vector_type(16))) _Float16 v16h;
typedef __attribute__((ext_vector_type(8)))  _Float16 v8h;
typedef __attribute__((ext_vector_type(4)))  _Float16 v4h;
typedef __attribute__((ext_vector_type(8)))  float    v8f;

#define NN 50000   // N_NODES; 50000/16 = 3125 = 5^5 row tiles
#define DF 128     // input / hidden feature dim
#define MT 5       // M tiles per block (3125 / 5 = 625 blocks)

// ---- in-degree: cnt[dst] += 1 ---------------------------------------------
__global__ void sage_degree(const long long* __restrict__ dst,
                            float* __restrict__ cnt, int E) {
    int e = blockIdx.x * blockDim.x + threadIdx.x;
    if (e < E) atomicAdd(&cnt[(int)dst[e]], 1.0f);
}

// ---- scatter-add: agg[dst] += feat_f16[src], one wave32 per edge ----------
__global__ void sage_scatter(const _Float16* __restrict__ feat,
                             const long long* __restrict__ src,
                             const long long* __restrict__ dst,
                             float* __restrict__ agg, int E) {
    int edge = blockIdx.x * (blockDim.x >> 5) + (threadIdx.x >> 5);
    if (edge >= E) return;
    int lane = threadIdx.x & 31;
    long long s = src[edge];
    long long d = dst[edge];
    const v4h v = *(const v4h*)(feat + (size_t)s * DF + lane * 4);
    float* op = agg + (size_t)d * DF + lane * 4;
    atomicAdd(op + 0, (float)v[0]);
    atomicAdd(op + 1, (float)v[1]);
    atomicAdd(op + 2, (float)v[2]);
    atomicAdd(op + 3, (float)v[3]);
}

// ---- weight prep: W[K][N] f32 -> WT[N][K] f16 (transpose + convert) -------
__global__ void sage_wprep(const float* __restrict__ W, _Float16* __restrict__ WT,
                           int K, int N) {
    int idx = blockIdx.x * blockDim.x + threadIdx.x;
    if (idx < K * N) {
        int k = idx / N, n = idx - k * N;
        WT[(size_t)n * K + k] = (_Float16)W[idx];
    }
}

// ---- operand prep: dstH[i] = f16(src[i] * (cnt ? 1/max(cnt[row],1) : 1)) --
__global__ void sage_cvt(const float* __restrict__ src,
                         const float* __restrict__ cnt,   // nullptr => scale 1
                         _Float16* __restrict__ dstH, int total4) {
    int i = blockIdx.x * blockDim.x + threadIdx.x;
    if (i >= total4) return;
    int base = i * 4;
    float s = 1.0f;
    if (cnt) s = 1.0f / fmaxf(cnt[base >> 7], 1.0f);
    float4 v = *(const float4*)(src + base);
    v4h o;
    o[0] = (_Float16)(v.x * s);
    o[1] = (_Float16)(v.y * s);
    o[2] = (_Float16)(v.z * s);
    o[3] = (_Float16)(v.w * s);
    *(v4h*)(dstH + base) = o;
}

// ---- fused SAGE layer: out = [relu]( aggH @ Wl + b + selfH @ Wr ) ---------
// ISA 7.12.2 fragment layouts (16-bit, wave32):
//   A 16x32 : lane holds row M = lane&15; half = lane>>4;
//             elem e<8 -> K = k0+8*half+e ; elem e>=8 -> K = k0+16+8*half+(e-8)
//             -> with f16 row-major operands: two contiguous v8h at +0, +16
//   B 32x16 : lane holds col N = lane&15; same K pattern over WT[N][K]
//   C 16x16 : vgpr i -> row M = i + 8*(lane>>4); col N = lane&15
template <int DOUT, bool RELU, typename OutT>
__global__ void sage_gemm(const _Float16* __restrict__ aggH,  // [NN, DF] f16, pre-scaled
                          const _Float16* __restrict__ selfH, // [NN, DF] f16
                          const _Float16* __restrict__ WlT,   // [DOUT, DF] f16
                          const _Float16* __restrict__ WrT,   // [DOUT, DF] f16
                          const float* __restrict__ bias,     // [DOUT] f32
                          OutT* __restrict__ out)             // [NN, DOUT]
{
    const int n0   = (threadIdx.x >> 5) * 16;
    const int lane = threadIdx.x & 31;
    const int r    = lane & 15;
    const int half = lane >> 4;

    // ---- hoisted B fragments: 4 k-steps x {Wl, Wr} (64 VGPRs) ----
    v16h bL[4], bR[4];
    {
        const _Float16* blRow = WlT + (size_t)(n0 + r) * DF + 8 * half;
        const _Float16* brRow = WrT + (size_t)(n0 + r) * DF + 8 * half;
#pragma unroll
        for (int ks = 0; ks < 4; ++ks) {
            v8h lo_l = *(const v8h*)(blRow + ks * 32);
            v8h hi_l = *(const v8h*)(blRow + ks * 32 + 16);
            v8h lo_r = *(const v8h*)(brRow + ks * 32);
            v8h hi_r = *(const v8h*)(brRow + ks * 32 + 16);
#pragma unroll
            for (int e = 0; e < 8; ++e) {
                bL[ks][e] = lo_l[e]; bL[ks][e + 8] = hi_l[e];
                bR[ks][e] = lo_r[e]; bR[ks][e + 8] = hi_r[e];
            }
        }
    }

    const float bv = bias[n0 + r];

#pragma unroll 1   // keep rolled: one tile in flight, no register spills
    for (int t = 0; t < MT; ++t) {
        const int m0 = (blockIdx.x * MT + t) * 16;
        const _Float16* aRow = aggH  + (size_t)(m0 + r) * DF + 8 * half;
        const _Float16* sRow = selfH + (size_t)(m0 + r) * DF + 8 * half;

        // Stage 1: issue ALL A-side loads for this tile into distinct
        // registers (16 x b128) so the compiler can clause them and use
        // partial loadcnt waits between the WMMAs below.
        v16h aN[4], aS[4];
#pragma unroll
        for (int ks = 0; ks < 4; ++ks) {
            v8h aLo = *(const v8h*)(aRow + ks * 32);
            v8h aHi = *(const v8h*)(aRow + ks * 32 + 16);
            v8h sLo = *(const v8h*)(sRow + ks * 32);
            v8h sHi = *(const v8h*)(sRow + ks * 32 + 16);
#pragma unroll
            for (int e = 0; e < 8; ++e) {
                aN[ks][e] = aLo[e]; aN[ks][e + 8] = aHi[e];
                aS[ks][e] = sLo[e]; aS[ks][e + 8] = sHi[e];
            }
        }

        // Stage 2: 8 chained WMMAs.
        v8f acc = {};
#pragma unroll
        for (int ks = 0; ks < 4; ++ks) {
            acc = __builtin_amdgcn_wmma_f32_16x16x32_f16(
                      false, aN[ks], false, bL[ks], (short)0, acc, false, false);
            acc = __builtin_amdgcn_wmma_f32_16x16x32_f16(
                      false, aS[ks], false, bR[ks], (short)0, acc, false, false);
        }

#pragma unroll
        for (int i = 0; i < 8; ++i) {
            float v = acc[i] + bv;
            if (RELU) v = fmaxf(v, 0.0f);
            out[(size_t)(m0 + i + 8 * half) * DOUT + (n0 + r)] = (OutT)v;
        }
    }
}

// ---------------------------------------------------------------------------
extern "C" void kernel_launch(void* const* d_in, const int* in_sizes, int n_in,
                              void* d_out, int out_size, void* d_ws, size_t ws_size,
                              hipStream_t stream) {
    const float*     x   = (const float*)d_in[0];
    const long long* ei  = (const long long*)d_in[1];  // int64 edge_index [2, E]
    const float*     W1l = (const float*)d_in[2];
    const float*     b1  = (const float*)d_in[3];
    const float*     W1r = (const float*)d_in[4];
    const float*     W2l = (const float*)d_in[5];
    const float*     b2  = (const float*)d_in[6];
    const float*     W2r = (const float*)d_in[7];
    float*           out = (float*)d_out;

    const int E = in_sizes[1] / 2;
    const long long* src = ei;
    const long long* dst = ei + E;

    // Workspace layout (256B aligned):
    //   cnt[NN] f32 | agg[NN*DF] f32 | aggH, xH, hH [NN*DF] f16 | weights f16
    char* ws = (char*)d_ws;
    const size_t cntBytes  = ((size_t)NN * sizeof(float) + 255) & ~(size_t)255;
    const size_t aggBytes  = (size_t)NN * DF * sizeof(float);      // 25.6MB
    const size_t halfBytes = (size_t)NN * DF * sizeof(_Float16);   // 12.8MB
    const size_t w1Bytes   = (size_t)DF * DF * sizeof(_Float16);   // 32KB
    const size_t w2Bytes   = (size_t)DF * 64 * sizeof(_Float16);   // 16KB
    float*    cnt  = (float*)ws;
    float*    agg  = (float*)(ws + cntBytes);
    _Float16* aggH = (_Float16*)(ws + cntBytes + aggBytes);
    _Float16* xH   = (_Float16*)((char*)aggH + halfBytes);
    _Float16* hH   = (_Float16*)((char*)xH + halfBytes);
    _Float16* W1lT = (_Float16*)((char*)hH + halfBytes);
    _Float16* W1rT = (_Float16*)((char*)W1lT + w1Bytes);
    _Float16* W2lT = (_Float16*)((char*)W1rT + w1Bytes);
    _Float16* W2rT = (_Float16*)((char*)W2lT + w2Bytes);

    const int blocks = NN / (16 * MT);       // 625, exact
    const int cvt4   = NN * DF / 4;          // 1.6M threads for cvt passes

    // ---- one-time per launch: operand prep (tiny vs. scatter cost) ----
    sage_wprep<<<(DF * DF + 255) / 256, 256, 0, stream>>>(W1l, W1lT, DF, DF);
    sage_wprep<<<(DF * DF + 255) / 256, 256, 0, stream>>>(W1r, W1rT, DF, DF);
    sage_wprep<<<(DF * 64 + 255) / 256, 256, 0, stream>>>(W2l, W2lT, DF, 64);
    sage_wprep<<<(DF * 64 + 255) / 256, 256, 0, stream>>>(W2r, W2rT, DF, 64);
    sage_cvt<<<(cvt4 + 255) / 256, 256, 0, stream>>>(x, nullptr, xH, cvt4);

    // ---- Layer 1 ----
    hipMemsetAsync(cnt, 0, (size_t)NN * sizeof(float), stream);
    hipMemsetAsync(agg, 0, aggBytes, stream);
    sage_degree<<<(E + 255) / 256, 256, 0, stream>>>(dst, cnt, E);
    sage_scatter<<<(E + 7) / 8, 256, 0, stream>>>(xH, src, dst, agg, E);
    sage_cvt<<<(cvt4 + 255) / 256, 256, 0, stream>>>(agg, cnt, aggH, cvt4);
    sage_gemm<128, true, _Float16><<<blocks, 256, 0, stream>>>(
        aggH, xH, W1lT, W1rT, b1, hH);

    // ---- Layer 2 ----
    hipMemsetAsync(agg, 0, aggBytes, stream);
    sage_scatter<<<(E + 7) / 8, 256, 0, stream>>>(hH, src, dst, agg, E);
    sage_cvt<<<(cvt4 + 255) / 256, 256, 0, stream>>>(agg, cnt, aggH, cvt4);
    sage_gemm<64, false, float><<<blocks, 128, 0, stream>>>(
        aggH, hH, W2lT, W2rT, b2, out);
}
